// AttentionBlock_7275674600243
// MI455X (gfx1250) — matrix operand order repaired and checked by reading, not compiled, not run on hardware
//
#include <hip/hip_runtime.h>

#define B_DIM   4
#define C_DIM   256
#define N_TOK   4096
#define NH_DIM  4
#define HD_DIM  64
#define OUT3C   768
#define CPG     32      // channels per group (256/8)

typedef __attribute__((ext_vector_type(16))) __bf16 bf16x16;
typedef __attribute__((ext_vector_type(8)))  __bf16 bf16x8;
typedef __attribute__((ext_vector_type(8)))  float  f32x8;
typedef __attribute__((ext_vector_type(4))) unsigned int u32x4;
typedef __attribute__((ext_vector_type(4))) int  i32x4;
typedef __attribute__((ext_vector_type(8))) int  i32x8;

#if __has_builtin(__builtin_amdgcn_tensor_load_to_lds)
#define USE_TDM 1
#else
#define USE_TDM 0
#endif

__device__ __forceinline__ __bf16 f2bf(float f) {
    unsigned u = __builtin_bit_cast(unsigned, f);
    unsigned r = u + 0x7FFFu + ((u >> 16) & 1u);   // round-to-nearest-even
    return __builtin_bit_cast(__bf16, (unsigned short)(r >> 16));
}

__device__ __forceinline__ f32x8 wmma_bf16(bf16x16 a, bf16x16 b, f32x8 c) {
    return __builtin_amdgcn_wmma_f32_16x16x32_bf16(false, a, false, b, (short)0, c,
                                                   false, false);
}

// A-fragment (16x32 bf16): row M=l16, elems[0..15]=K 16*half..16*half+15 (contiguous)
__device__ __forceinline__ bf16x16 load_a_frag(const __bf16* row, int half) {
    bf16x8 lo = *reinterpret_cast<const bf16x8*>(row + 16 * half);
    bf16x8 hi = *reinterpret_cast<const bf16x8*>(row + 16 * half + 8);
    bf16x16 a;
#pragma unroll
    for (int i = 0; i < 8; ++i) { a[i] = lo[i]; a[i + 8] = hi[i]; }
    return a;
}

// B-fragment (32x16 bf16): lane holds column l16; 16 contiguous K at 16*half
__device__ __forceinline__ bf16x16 load_b_frag(const __bf16* colrow, int half) {
    return *reinterpret_cast<const bf16x16*>(colrow + 16 * half);
}

#if USE_TDM
// 2D Tensor-DMA descriptor per CDNA5 ISA ch.8: group0 = count/lds/global/type,
// group1 = data_size + tensor dims + tile dims + 48-bit dim0 stride.
__device__ __forceinline__ void tdm_load_2d(unsigned lds_addr, const void* gaddr,
                                            unsigned tile_w, unsigned tile_h,
                                            unsigned tensor_w, unsigned tensor_h,
                                            unsigned long long stride0) {
    unsigned long long ga = (unsigned long long)(size_t)gaddr;
    u32x4 g0;
    g0[0] = 1u;                                            // count=1, user descriptor
    g0[1] = lds_addr;                                      // LDS byte address
    g0[2] = (unsigned)(ga & 0xFFFFFFFFu);                  // global_addr[31:0]
    g0[3] = (unsigned)((ga >> 32) & 0x01FFFFFFu) | (2u << 30);  // [120:64] + type=2
    i32x8 g1;
    g1[0] = (int)(1u << 16);                               // data_size = 2 bytes
    g1[1] = (int)((tensor_w & 0xFFFFu) << 16);             // tensor_dim0[15:0]
    g1[2] = (int)(((tensor_w >> 16) & 0xFFFFu) | ((tensor_h & 0xFFFFu) << 16));
    g1[3] = (int)(((tensor_h >> 16) & 0xFFFFu) | ((tile_w & 0xFFFFu) << 16));
    g1[4] = (int)(tile_h & 0xFFFFu);                       // tile_dim1; tile_dim2=0
    g1[5] = (int)(stride0 & 0xFFFFFFFFull);                // tensor_dim0_stride[31:0]
    g1[6] = (int)((stride0 >> 32) & 0xFFFFull);            // stride[47:32]; dim1_stride=0
    g1[7] = 0;
    i32x4 z4 = {0, 0, 0, 0};
#if __clang_major__ >= 23
    i32x8 z8 = {0, 0, 0, 0, 0, 0, 0, 0};
    __builtin_amdgcn_tensor_load_to_lds(g0, g1, z4, z4, z8, 0);
#else
    __builtin_amdgcn_tensor_load_to_lds(g0, g1, z4, z4, 0);
#endif
}
#endif

// ---------------------------------------------------------------- f32 -> bf16
__global__ __launch_bounds__(256) void cvt_bf16_kernel(const float* __restrict__ s,
                                                       __bf16* __restrict__ d, int n) {
    int i = blockIdx.x * 256 + threadIdx.x;
    if (i < n) d[i] = f2bf(s[i]);
}

// ---------------------------------------------------------------- GN stats
__global__ __launch_bounds__(256) void gn_stats_kernel(const float* __restrict__ x,
                                                       float* __restrict__ stats) {
    const int bg  = blockIdx.x;                       // b*8 + g
    const int tid = threadIdx.x;
    const size_t base = (size_t)bg * CPG * N_TOK;     // groups are contiguous
    const float4* p = reinterpret_cast<const float4*>(x + base);
    const int nvec = CPG * N_TOK / 4;
    float s = 0.f, ss = 0.f;
    for (int i = tid; i < nvec; i += 256) {
        float4 v = p[i];
        s  += v.x + v.y + v.z + v.w;
        ss += v.x * v.x + v.y * v.y + v.z * v.z + v.w * v.w;
    }
#pragma unroll
    for (int off = 1; off < 32; off <<= 1) {
        s  += __shfl_xor(s, off);
        ss += __shfl_xor(ss, off);
    }
    __shared__ float red[16];
    const int wid = tid >> 5;
    if ((tid & 31) == 0) { red[wid * 2] = s; red[wid * 2 + 1] = ss; }
    __syncthreads();
    if (tid == 0) {
        float S = 0.f, SS = 0.f;
        for (int w = 0; w < 8; ++w) { S += red[w * 2]; SS += red[w * 2 + 1]; }
        const float inv  = 1.0f / (float)(CPG * N_TOK);
        const float mean = S * inv;
        const float var  = SS * inv - mean * mean;
        stats[bg * 2]     = mean;
        stats[bg * 2 + 1] = rsqrtf(var + 1e-5f);
    }
}

// ------------------------------------------------- GN apply + (c,n)->(n,c) bf16
__global__ __launch_bounds__(256) void gn_pack_kernel(const float* __restrict__ x,
                                                      const float* __restrict__ w,
                                                      const float* __restrict__ bia,
                                                      const float* __restrict__ stats,
                                                      __bf16* __restrict__ hT) {
    __shared__ float tile[32][33];
    const int tx = threadIdx.x, ty = threadIdx.y;
    const int n0 = blockIdx.x * 32, c0 = blockIdx.y * 32, b = blockIdx.z;
#pragma unroll
    for (int k = 0; k < 4; ++k) {
        const int c = c0 + ty + 8 * k;
        const int g = c >> 5;
        const float mean = stats[(b * 8 + g) * 2];
        const float rstd = stats[(b * 8 + g) * 2 + 1];
        const float v = x[((size_t)b * C_DIM + c) * N_TOK + n0 + tx];
        tile[ty + 8 * k][tx] = (v - mean) * rstd * w[c] + bia[c];
    }
    __syncthreads();
#pragma unroll
    for (int k = 0; k < 4; ++k) {
        const int n = n0 + ty + 8 * k;
        hT[((size_t)b * N_TOK + n) * C_DIM + c0 + tx] = f2bf(tile[tx][ty + 8 * k]);
    }
}

// ---------------------- QKV GEMM: one wave = 16x64 macro-tile (A reused 4x)
__global__ __launch_bounds__(32) void qkv_gemm_kernel(const __bf16* __restrict__ h,
                                                      const __bf16* __restrict__ wq,
                                                      const float* __restrict__ qb,
                                                      __bf16* __restrict__ Q,
                                                      __bf16* __restrict__ K,
                                                      __bf16* __restrict__ Vt) {
    const int lane = threadIdx.x, l16 = lane & 15, half = lane >> 4;
    const int n0 = blockIdx.x * 16, o0 = blockIdx.y * 64, b = blockIdx.z;
    const __bf16* arow = h + ((size_t)b * N_TOK + n0 + l16) * C_DIM;
    const __bf16* brow[4];
#pragma unroll
    for (int j = 0; j < 4; ++j) brow[j] = wq + (size_t)(o0 + 16 * j + l16) * C_DIM;
    const f32x8 zf = {};
    f32x8 acc[4] = {zf, zf, zf, zf};
#pragma unroll
    for (int k0 = 0; k0 < C_DIM; k0 += 32) {
        bf16x16 a = load_a_frag(arow + k0, half);
#pragma unroll
        for (int j = 0; j < 4; ++j)
            acc[j] = wmma_bf16(a, load_b_frag(brow[j] + k0, half), acc[j]);
    }
    const int region = o0 >> 8;                 // uniform: 0=Q, 1=K, 2=V
    if (region == 0) {
#pragma unroll
        for (int j = 0; j < 4; ++j) {
            const int o = o0 + 16 * j + l16, head = (o & 255) >> 6, d = o & 63;
            const float bias = qb[o];
            __bf16* dst = Q + ((size_t)b * NH_DIM + head) * N_TOK * HD_DIM;
#pragma unroll
            for (int r = 0; r < 8; ++r) {
                const int n = n0 + r + 8 * half;
                dst[(size_t)n * HD_DIM + d] = f2bf((acc[j][r] + bias) * 0.125f);
            }
        }
    } else if (region == 1) {
#pragma unroll
        for (int j = 0; j < 4; ++j) {
            const int o = o0 + 16 * j + l16, head = (o & 255) >> 6, d = o & 63;
            const float bias = qb[o];
            __bf16* dst = K + ((size_t)b * NH_DIM + head) * N_TOK * HD_DIM;
#pragma unroll
            for (int r = 0; r < 8; ++r) {
                const int n = n0 + r + 8 * half;
                dst[(size_t)n * HD_DIM + d] = f2bf(acc[j][r] + bias);
            }
        }
    } else {                                    // V stored transposed (d, m)
#pragma unroll
        for (int j = 0; j < 4; ++j) {
            const int o = o0 + 16 * j + l16, head = (o & 255) >> 6, d = o & 63;
            const float bias = qb[o];
            __bf16* dst = Vt + (((size_t)b * NH_DIM + head) * HD_DIM + d) * N_TOK;
#pragma unroll
            for (int r = 0; r < 8; ++r) dst[n0 + r + 8 * half] = f2bf(acc[j][r] + bias);
        }
    }
}

// ------------------------------------------------- flash attention (TDM-staged K/V)
__global__ __launch_bounds__(256) void attn_kernel(const __bf16* __restrict__ q,
                                                   const __bf16* __restrict__ k,
                                                   const __bf16* __restrict__ vt,
                                                   __bf16* __restrict__ oT) {
    __shared__ __bf16 kTile[2][32][64] __attribute__((aligned(64)));  // (m, d)
    __shared__ __bf16 vTile[2][64][32] __attribute__((aligned(64)));  // (d, m)
    __shared__ __bf16 lds_p[8][16][32] __attribute__((aligned(64)));
    const int tid = threadIdx.x, wid = tid >> 5, lane = tid & 31;
    const int l16 = lane & 15, half = lane >> 4;
    const int b = blockIdx.z, head = blockIdx.y;
    const int n0 = (blockIdx.x * 8 + wid) * 16;
    const size_t base = ((size_t)b * NH_DIM + head) * (size_t)N_TOK * HD_DIM;
    const __bf16* qp = q  + base;
    const __bf16* kp = k  + base;
    const __bf16* vp = vt + base;   // (d, m): base + d*N + m

    bf16x16 qa[2];
#pragma unroll
    for (int t = 0; t < 2; ++t)
        qa[t] = load_a_frag(qp + (size_t)(n0 + l16) * HD_DIM + 32 * t, half);

    float m_i[8], l_i[8];
    f32x8 accO[4];
    const f32x8 zf = {};
#pragma unroll
    for (int r = 0; r < 8; ++r) { m_i[r] = -1e30f; l_i[r] = 0.f; }
#pragma unroll
    for (int j = 0; j < 4; ++j) accO[j] = zf;

    // --- stage first K/V tile (double-buffered) ---
#if USE_TDM
    if (wid == 0) {
        tdm_load_2d((unsigned)(size_t)&kTile[0][0][0], kp, HD_DIM, 32,
                    HD_DIM, N_TOK, HD_DIM);
        tdm_load_2d((unsigned)(size_t)&vTile[0][0][0], vp, 32, HD_DIM,
                    N_TOK, HD_DIM, N_TOK);
        __builtin_amdgcn_s_wait_tensorcnt(0);
    }
#else
    *(bf16x8*)&kTile[0][tid >> 3][(tid & 7) * 8] =
        *(const bf16x8*)(kp + (size_t)(tid >> 3) * HD_DIM + (tid & 7) * 8);
    *(bf16x8*)&vTile[0][tid >> 2][(tid & 3) * 8] =
        *(const bf16x8*)(vp + (size_t)(tid >> 2) * N_TOK + (tid & 3) * 8);
#endif
    __syncthreads();

    int buf = 0;
    for (int m0 = 0; m0 < N_TOK; m0 += 32) {
        // prefetch next tile into the other buffer while computing
        if (m0 + 32 < N_TOK) {
            const int nb = buf ^ 1, mn = m0 + 32;
#if USE_TDM
            if (wid == 0) {
                tdm_load_2d((unsigned)(size_t)&kTile[nb][0][0],
                            kp + (size_t)mn * HD_DIM, HD_DIM, 32, HD_DIM, N_TOK, HD_DIM);
                tdm_load_2d((unsigned)(size_t)&vTile[nb][0][0],
                            vp + mn, 32, HD_DIM, N_TOK, HD_DIM, N_TOK);
            }
#else
            *(bf16x8*)&kTile[nb][tid >> 3][(tid & 7) * 8] =
                *(const bf16x8*)(kp + (size_t)(mn + (tid >> 3)) * HD_DIM + (tid & 7) * 8);
            *(bf16x8*)&vTile[nb][tid >> 2][(tid & 3) * 8] =
                *(const bf16x8*)(vp + (size_t)(tid >> 2) * N_TOK + mn + (tid & 3) * 8);
#endif
        }
        // S = Q * K^T  (B-fragments straight out of LDS)
        f32x8 s[2];
#pragma unroll
        for (int t = 0; t < 2; ++t) {
            const __bf16* krow = &kTile[buf][16 * t + l16][16 * half];
            bf16x16 kb0 = *(const bf16x16*)krow;          // d = 0..31
            bf16x16 kb1 = *(const bf16x16*)(krow + 32);   // d = 32..63
            f32x8 st = zf;
            st = wmma_bf16(qa[0], kb0, st);
            st = wmma_bf16(qa[1], kb1, st);
            s[t] = st;
        }
        // online softmax (row = r + 8*half, replicated across 16-lane half)
#pragma unroll
        for (int r = 0; r < 8; ++r) {
            float s0 = s[0][r], s1 = s[1][r];
            float mx = fmaxf(s0, s1);
#pragma unroll
            for (int off = 1; off < 16; off <<= 1) mx = fmaxf(mx, __shfl_xor(mx, off));
            const float mnew = fmaxf(m_i[r], mx);
            const float corr = __expf(m_i[r] - mnew);
            m_i[r] = mnew;
            const float p0 = __expf(s0 - mnew), p1 = __expf(s1 - mnew);
            float rs = p0 + p1;
#pragma unroll
            for (int off = 1; off < 16; off <<= 1) rs += __shfl_xor(rs, off);
            l_i[r] = l_i[r] * corr + rs;
#pragma unroll
            for (int j = 0; j < 4; ++j) accO[j][r] *= corr;
            lds_p[wid][r + 8 * half][l16]      = f2bf(p0);
            lds_p[wid][r + 8 * half][16 + l16] = f2bf(p1);
        }
        asm volatile("s_wait_dscnt 0" ::: "memory");
        // reread P in A-fragment layout
        const bf16x8* prow = reinterpret_cast<const bf16x8*>(&lds_p[wid][l16][0]);
        bf16x8 plo = prow[2 * half], phi = prow[2 * half + 1];
        bf16x16 pa;
#pragma unroll
        for (int i = 0; i < 8; ++i) { pa[i] = plo[i]; pa[i + 8] = phi[i]; }
        // O += P * V
#pragma unroll
        for (int j = 0; j < 4; ++j) {
            bf16x16 vb = *(const bf16x16*)&vTile[buf][16 * j + l16][16 * half];
            accO[j] = wmma_bf16(pa, vb, accO[j]);
        }
#if USE_TDM
        if (wid == 0) __builtin_amdgcn_s_wait_tensorcnt(0);
#endif
        __syncthreads();
        buf ^= 1;
    }
#pragma unroll
    for (int r = 0; r < 8; ++r) {
        const float inv = 1.0f / l_i[r];
        const int n = n0 + r + 8 * half;
        __bf16* dst = oT + ((size_t)b * N_TOK + n) * C_DIM + head * HD_DIM;
#pragma unroll
        for (int j = 0; j < 4; ++j) dst[16 * j + l16] = f2bf(accO[j][r] * inv);
    }
}

// ---------------------- proj GEMM: 16x64 macro-tile + bias + residual
__global__ __launch_bounds__(32) void proj_gemm_kernel(const __bf16* __restrict__ oT,
                                                       const __bf16* __restrict__ wp,
                                                       const float* __restrict__ pb,
                                                       const float* __restrict__ x,
                                                       float* __restrict__ out) {
    const int lane = threadIdx.x, l16 = lane & 15, half = lane >> 4;
    const int n0 = blockIdx.x * 16, co0 = blockIdx.y * 64, b = blockIdx.z;
    const __bf16* arow = oT + ((size_t)b * N_TOK + n0 + l16) * C_DIM;
    const __bf16* brow[4];
#pragma unroll
    for (int j = 0; j < 4; ++j) brow[j] = wp + (size_t)(co0 + 16 * j + l16) * C_DIM;
    const f32x8 zf = {};
    f32x8 acc[4] = {zf, zf, zf, zf};
#pragma unroll
    for (int k0 = 0; k0 < C_DIM; k0 += 32) {
        bf16x16 a = load_a_frag(arow + k0, half);
#pragma unroll
        for (int j = 0; j < 4; ++j)
            acc[j] = wmma_bf16(a, load_b_frag(brow[j] + k0, half), acc[j]);
    }
#pragma unroll
    for (int j = 0; j < 4; ++j) {
        const int co = co0 + 16 * j + l16;
        const float bias = pb[co];
        const size_t ob = ((size_t)b * C_DIM + co) * N_TOK;
#pragma unroll
        for (int r = 0; r < 8; ++r) {
            const int n = n0 + r + 8 * half;
            out[ob + n] = x[ob + n] + acc[j][r] + bias;
        }
    }
}

// ----------------------------------------------------------------------------
extern "C" void kernel_launch(void* const* d_in, const int* in_sizes, int n_in,
                              void* d_out, int out_size, void* d_ws, size_t ws_size,
                              hipStream_t stream) {
    (void)in_sizes; (void)n_in; (void)out_size; (void)ws_size;
    const float* x      = (const float*)d_in[0];
    const float* norm_w = (const float*)d_in[1];
    const float* norm_b = (const float*)d_in[2];
    const float* qkv_w  = (const float*)d_in[3];
    const float* qkv_b  = (const float*)d_in[4];
    const float* proj_w = (const float*)d_in[5];
    const float* proj_b = (const float*)d_in[6];
    float* out = (float*)d_out;

    char* ws = (char*)d_ws;
    float* stats = (float*)ws;
    size_t off = 1024;
    __bf16* wq_bf = (__bf16*)(ws + off); off += (size_t)OUT3C * C_DIM * 2;
    __bf16* wp_bf = (__bf16*)(ws + off); off += (size_t)C_DIM * C_DIM * 2;
    __bf16* hT    = (__bf16*)(ws + off); off += (size_t)B_DIM * N_TOK * C_DIM * 2;
    __bf16* Q     = (__bf16*)(ws + off); off += (size_t)B_DIM * N_TOK * C_DIM * 2;
    __bf16* K     = (__bf16*)(ws + off); off += (size_t)B_DIM * N_TOK * C_DIM * 2;
    __bf16* Vt    = (__bf16*)(ws + off); off += (size_t)B_DIM * N_TOK * C_DIM * 2;
    __bf16* oT    = (__bf16*)(ws + off);

    cvt_bf16_kernel<<<dim3((OUT3C * C_DIM + 255) / 256), dim3(256), 0, stream>>>(
        qkv_w, wq_bf, OUT3C * C_DIM);
    cvt_bf16_kernel<<<dim3((C_DIM * C_DIM + 255) / 256), dim3(256), 0, stream>>>(
        proj_w, wp_bf, C_DIM * C_DIM);
    gn_stats_kernel<<<dim3(B_DIM * 8), dim3(256), 0, stream>>>(x, stats);
    gn_pack_kernel<<<dim3(N_TOK / 32, C_DIM / 32, B_DIM), dim3(32, 8), 0, stream>>>(
        x, norm_w, norm_b, stats, hT);
    qkv_gemm_kernel<<<dim3(N_TOK / 16, OUT3C / 64, B_DIM), dim3(32), 0, stream>>>(
        hT, wq_bf, qkv_b, Q, K, Vt);
    attn_kernel<<<dim3(N_TOK / 128, NH_DIM, B_DIM), dim3(256), 0, stream>>>(Q, K, Vt, oT);
    proj_gemm_kernel<<<dim3(N_TOK / 16, C_DIM / 64, B_DIM), dim3(32), 0, stream>>>(
        oT, wp_bf, proj_b, x, out);
}
